// MultiHeadAttention_60352880443845
// MI455X (gfx1250) — compile-verified, matrix operand order checked by reading, not run on hardware
//
#include <hip/hip_runtime.h>

// ---------------------------------------------------------------------------
// MHA forward for MI455X (gfx1250, wave32, WMMA).
// Pipeline: f32->bf16 convert | QKV GEMM (bf16 wmma, async-LDS double buffer) |
//           flash attention | out-proj GEMM (+bias, f32 out).
// ---------------------------------------------------------------------------

#ifndef __has_builtin
#define __has_builtin(x) 0
#endif
#if __has_builtin(__builtin_amdgcn_global_load_async_to_lds_b128) && \
    __has_builtin(__builtin_amdgcn_s_wait_asynccnt)
#define USE_ASYNC_LDS 1
#else
#define USE_ASYNC_LDS 0
#endif

typedef __attribute__((ext_vector_type(16))) __bf16          v16bf;
typedef __attribute__((ext_vector_type(8)))  float           v8f;
typedef __attribute__((ext_vector_type(8)))  unsigned short  u16x8;
typedef __attribute__((ext_vector_type(4)))  unsigned short  u16x4;
typedef __attribute__((ext_vector_type(4)))  float           f32x4;
typedef __attribute__((ext_vector_type(4)))  int             i32x4;

#define GLOBAL_AS __attribute__((address_space(1)))
#define LDS_AS    __attribute__((address_space(3)))

static constexpr int BATCH = 4;
static constexpr int SEQ   = 2048;
static constexpr int DMODEL= 1024;
static constexpr int NHEAD = 16;
static constexpr int HDIM  = 64;
static constexpr int MROWS = BATCH * SEQ;       // 8192
static constexpr int QKVN  = 3 * DMODEL;        // 3072

__device__ __forceinline__ unsigned short f32_to_bf16_bits(float f) {
  union { float f; unsigned u; } x; x.f = f;
  unsigned r = x.u + 0x7FFFu + ((x.u >> 16) & 1u);   // round-to-nearest-even
  return (unsigned short)(r >> 16);
}
__device__ __forceinline__ float bf16_bits_to_f32(unsigned short h) {
  union { unsigned u; float f; } x; x.u = ((unsigned)h) << 16;
  return x.f;
}

// 16-byte global -> LDS copy: async (ASYNCcnt) when available, else via VGPRs.
__device__ __forceinline__ void copy16_to_lds(const unsigned short* g,
                                              unsigned short* l) {
#if USE_ASYNC_LDS
  __builtin_amdgcn_global_load_async_to_lds_b128(
      (GLOBAL_AS i32x4*)g, (LDS_AS i32x4*)l, 0, 0);
#else
  *(u16x8*)l = *(const u16x8*)g;
#endif
}
__device__ __forceinline__ void wait_async_le8() {
#if USE_ASYNC_LDS
  __builtin_amdgcn_s_wait_asynccnt(8);
#endif
}
__device__ __forceinline__ void wait_async_all() {
#if USE_ASYNC_LDS
  __builtin_amdgcn_s_wait_asynccnt(0);
#endif
}

// Load one 16x32 bf16 fragment (A layout; B uses the same pattern on W rows).
// Lanes 0-15: row = r, cols [c0..c0+7] and [c0+16..c0+23]
// Lanes 16-31: row = r, cols [c0+8..c0+15] and [c0+24..c0+31]
template <int LD>
__device__ __forceinline__ v16bf load_frag(const unsigned short* base,
                                           int row0, int col0, int lane) {
  const int half = lane >> 4, r = lane & 15;
  const unsigned short* p = base + (row0 + r) * LD + col0 + half * 8;
  union { v16bf v; u16x8 h[2]; } f;
  f.h[0] = *(const u16x8*)(p);
  f.h[1] = *(const u16x8*)(p + 16);
  return f.v;
}

__device__ __forceinline__ float redmax16(float x) {
  x = fmaxf(x, __shfl_xor(x, 1, 32));
  x = fmaxf(x, __shfl_xor(x, 2, 32));
  x = fmaxf(x, __shfl_xor(x, 4, 32));
  x = fmaxf(x, __shfl_xor(x, 8, 32));
  return x;
}
__device__ __forceinline__ float redsum16(float x) {
  x += __shfl_xor(x, 1, 32);
  x += __shfl_xor(x, 2, 32);
  x += __shfl_xor(x, 4, 32);
  x += __shfl_xor(x, 8, 32);
  return x;
}

// ---------------------------------------------------------------------------
// Kernel 0: elementwise f32 -> bf16 (vectorized x4)
// ---------------------------------------------------------------------------
__global__ __launch_bounds__(256)
void cvt_f32_bf16(const float* __restrict__ in, unsigned short* __restrict__ out,
                  int n4) {
  int i = blockIdx.x * 256 + threadIdx.x;
  if (i >= n4) return;
  f32x4 v = *(const f32x4*)(in + (size_t)i * 4);
  u16x4 o;
  o.x = f32_to_bf16_bits(v.x);
  o.y = f32_to_bf16_bits(v.y);
  o.z = f32_to_bf16_bits(v.z);
  o.w = f32_to_bf16_bits(v.w);
  *(u16x4*)(out + (size_t)i * 4) = o;
}

// ---------------------------------------------------------------------------
// Kernel 1/3: C[M,N] = A[M,K] (bf16, row-major) x W[N,K]^T (bf16, row-major)
// 128x128x64 block tile, 8 waves (4x2), wave tile 32x64 -> 8 WMMA accumulators.
// Double-buffered LDS; next tile streamed with async global->LDS copies that
// overlap the current tile's 16 WMMAs.
// ---------------------------------------------------------------------------
template <bool OUT_BF16, bool BIAS>
__global__ __launch_bounds__(256)
void gemm_xwt(const unsigned short* __restrict__ A,
              const unsigned short* __restrict__ W,
              const float* __restrict__ bias,
              void* __restrict__ Cout,
              int M, int N, int K) {
  constexpr int BM = 128, BN = 128, BK = 64, LDT = BK + 8;  // pad: 144B stride
  __shared__ __align__(16) unsigned short As[2][BM * LDT];
  __shared__ __align__(16) unsigned short Ws[2][BN * LDT];

  const int tid  = threadIdx.x;
  const int lane = tid & 31, wid = tid >> 5;
  const int wm = wid & 3, wn = wid >> 2;                    // 4 x 2 wave grid
  const int bm = blockIdx.y * BM, bn = blockIdx.x * BN;

  // per-thread tile-stage: 4 x 16B of A + 4 x 16B of W = 8 async ops
  auto stage = [&](int k0, int buf) {
#pragma unroll
    for (int c = tid; c < BM * BK / 8; c += 256) {
      int row = c >> 3, seg = c & 7;
      copy16_to_lds(A + (size_t)(bm + row) * K + k0 + seg * 8,
                    &As[buf][row * LDT + seg * 8]);
    }
#pragma unroll
    for (int c = tid; c < BN * BK / 8; c += 256) {
      int row = c >> 3, seg = c & 7;
      copy16_to_lds(W + (size_t)(bn + row) * K + k0 + seg * 8,
                    &Ws[buf][row * LDT + seg * 8]);
    }
    if (k0 + BK < K) {                                      // prefetch t+2
      __builtin_prefetch(A + (size_t)(bm + (tid >> 1)) * K + k0 + BK, 0, 1);
      __builtin_prefetch(W + (size_t)(bn + (tid >> 1)) * K + k0 + BK, 0, 1);
    }
  };

  v8f acc[2][4] = {};
  const int nTiles = K / BK;
  stage(0, 0);

  for (int t = 0; t < nTiles; ++t) {
    const int cur = t & 1;
    if (t + 1 < nTiles) {
      stage((t + 1) * BK, cur ^ 1);   // overlaps compute below
      wait_async_le8();               // current tile (older 8 ops) complete
    } else {
      wait_async_all();
    }
    __syncthreads();

#pragma unroll
    for (int kk = 0; kk < BK; kk += 32) {
      v16bf af[2], bfr[4];
      af[0] = load_frag<LDT>(&As[cur][0], wm * 32,      kk, lane);
      af[1] = load_frag<LDT>(&As[cur][0], wm * 32 + 16, kk, lane);
#pragma unroll
      for (int j = 0; j < 4; j++)
        bfr[j] = load_frag<LDT>(&Ws[cur][0], wn * 64 + j * 16, kk, lane);
#pragma unroll
      for (int i = 0; i < 2; i++)
#pragma unroll
        for (int j = 0; j < 4; j++)
          acc[i][j] = __builtin_amdgcn_wmma_f32_16x16x32_bf16(
              false, af[i], false, bfr[j], (short)0, acc[i][j], false, false);
    }
    __syncthreads();   // everyone done reading `cur` before it is re-staged
  }

  const int half = lane >> 4, nlo = lane & 15;
#pragma unroll
  for (int i = 0; i < 2; i++) {
#pragma unroll
    for (int j = 0; j < 4; j++) {
      const int col = bn + wn * 64 + j * 16 + nlo;
#pragma unroll
      for (int r = 0; r < 8; r++) {
        const int row = bm + wm * 32 + i * 16 + r + 8 * half;
        float v = acc[i][j][r];
        if (BIAS) v += bias[col];
        if (OUT_BF16)
          ((unsigned short*)Cout)[(size_t)row * N + col] = f32_to_bf16_bits(v);
        else
          ((float*)Cout)[(size_t)row * N + col] = v;
      }
    }
  }
}

// ---------------------------------------------------------------------------
// Kernel 2: flash attention. One workgroup per (b, h, 128-row q block).
// qkv: [B*S, 3072] bf16 (Q | K | V, head h at col h*64). out: [B*S, 1024] bf16.
// ---------------------------------------------------------------------------
__global__ __launch_bounds__(256)
void attn_kernel(const unsigned short* __restrict__ qkv,
                 unsigned short* __restrict__ out) {
  constexpr int LDQ = 72;                                   // 64 + 8 pad
  __shared__ __align__(16) unsigned short Qs[128 * LDQ];
  __shared__ __align__(16) unsigned short Ks[64 * LDQ];
  __shared__ __align__(16) unsigned short Vt[64 * LDQ];     // transposed V
  __shared__ __align__(16) unsigned short Ps[128 * LDQ];

  const int qblk = blockIdx.x, h = blockIdx.y, b = blockIdx.z;
  const int tid = threadIdx.x, lane = tid & 31, wid = tid >> 5;
  const int half = lane >> 4, nlo = lane & 15;

  const size_t rowbase = (size_t)b * SEQ;
  const unsigned short* Qg = qkv + (rowbase + qblk * 128) * QKVN + h * HDIM;
  const unsigned short* Kg = qkv + rowbase * QKVN + DMODEL     + h * HDIM;
  const unsigned short* Vg = qkv + rowbase * QKVN + 2 * DMODEL + h * HDIM;

  // Q block, pre-scaled by 1/sqrt(64) = 0.125
#pragma unroll
  for (int c = tid; c < 1024; c += 256) {                   // 128*64/8 chunks
    int row = c >> 3, seg = c & 7;
    u16x8 d = *(const u16x8*)(Qg + (size_t)row * QKVN + seg * 8);
    u16x8 o;
#pragma unroll
    for (int j = 0; j < 8; j++)
      o[j] = f32_to_bf16_bits(bf16_bits_to_f32(d[j]) * 0.125f);
    *(u16x8*)(Qs + row * LDQ + seg * 8) = o;
  }

  float mrow[8], lrow[8];
  v8f acco[4] = {};
#pragma unroll
  for (int r = 0; r < 8; r++) { mrow[r] = -1.0e30f; lrow[r] = 0.0f; }

  for (int kb = 0; kb < SEQ; kb += 64) {
    __syncthreads();
#pragma unroll
    for (int c = tid; c < 512; c += 256) {                  // K rows (async)
      int row = c >> 3, seg = c & 7;
      copy16_to_lds(Kg + (size_t)(kb + row) * QKVN + seg * 8,
                    &Ks[row * LDQ + seg * 8]);
    }
#pragma unroll
    for (int c = tid; c < 512; c += 256) {                  // V, transposed
      int row = c >> 3, seg = c & 7;
      u16x8 d = *(const u16x8*)(Vg + (size_t)(kb + row) * QKVN + seg * 8);
#pragma unroll
      for (int j = 0; j < 8; j++)
        Vt[(seg * 8 + j) * LDQ + row] = d[j];
    }
    wait_async_all();
    __syncthreads();

    // S = Q(16x64) x K^T(64x64), per wave
    v8f accs[4] = {};
#pragma unroll
    for (int kk = 0; kk < 64; kk += 32) {
      v16bf qf = load_frag<LDQ>(Qs, wid * 16, kk, lane);
#pragma unroll
      for (int j = 0; j < 4; j++) {
        v16bf kf = load_frag<LDQ>(Ks, j * 16, kk, lane);
        accs[j] = __builtin_amdgcn_wmma_f32_16x16x32_bf16(
            false, qf, false, kf, (short)0, accs[j], false, false);
      }
    }

    // online softmax, rows striped across lanes 0-15 / 16-31
#pragma unroll
    for (int r = 0; r < 8; r++) {
      float mx = accs[0][r];
      mx = fmaxf(mx, accs[1][r]);
      mx = fmaxf(mx, accs[2][r]);
      mx = fmaxf(mx, accs[3][r]);
      mx = redmax16(mx);
      float mnew  = fmaxf(mrow[r], mx);
      float alpha = __expf(mrow[r] - mnew);
      float psum  = 0.0f;
#pragma unroll
      for (int j = 0; j < 4; j++) {
        float p = __expf(accs[j][r] - mnew);
        accs[j][r] = p;
        psum += p;
      }
      psum = redsum16(psum);
      lrow[r] = lrow[r] * alpha + psum;
      mrow[r] = mnew;
#pragma unroll
      for (int j = 0; j < 4; j++) acco[j][r] *= alpha;
    }

    // stage P (C-layout -> LDS row-major) for the PV WMMA
#pragma unroll
    for (int r = 0; r < 8; r++) {
      const int prow = wid * 16 + r + 8 * half;
#pragma unroll
      for (int j = 0; j < 4; j++)
        Ps[prow * LDQ + j * 16 + nlo] = f32_to_bf16_bits(accs[j][r]);
    }
    __syncthreads();

    // O += P(16x64) x V(64x64); Vt rows give contiguous B fragments
#pragma unroll
    for (int kk = 0; kk < 64; kk += 32) {
      v16bf pf = load_frag<LDQ>(Ps, wid * 16, kk, lane);
#pragma unroll
      for (int j = 0; j < 4; j++) {
        v16bf vf = load_frag<LDQ>(Vt, j * 16, kk, lane);
        acco[j] = __builtin_amdgcn_wmma_f32_16x16x32_bf16(
            false, pf, false, vf, (short)0, acco[j], false, false);
      }
    }
  }

  // normalize + write to [B*S, 1024] (cols h*64 + d)
#pragma unroll
  for (int r = 0; r < 8; r++) {
    const size_t orow = rowbase + qblk * 128 + wid * 16 + r + 8 * half;
    const float inv = 1.0f / lrow[r];
#pragma unroll
    for (int j = 0; j < 4; j++) {
      const int col = h * HDIM + j * 16 + nlo;
      out[orow * DMODEL + col] = f32_to_bf16_bits(acco[j][r] * inv);
    }
  }
}

// ---------------------------------------------------------------------------
extern "C" void kernel_launch(void* const* d_in, const int* in_sizes, int n_in,
                              void* d_out, int out_size, void* d_ws, size_t ws_size,
                              hipStream_t stream) {
  const float* query = (const float*)d_in[0];
  // d_in[1] (key) and d_in[2] (value) are unused by the reference module.
  const float* w_qkv = (const float*)d_in[3];
  const float* w_out = (const float*)d_in[4];
  const float* b_out = (const float*)d_in[5];
  float* out = (float*)d_out;

  unsigned short* Xbf    = (unsigned short*)d_ws;
  unsigned short* Wqkvbf = Xbf    + (size_t)MROWS * DMODEL;        // 8192*1024
  unsigned short* Woutbf = Wqkvbf + (size_t)QKVN * DMODEL;         // 3072*1024
  unsigned short* QKVbf  = Woutbf + (size_t)DMODEL * DMODEL;       // 1024*1024
  unsigned short* AObf   = QKVbf  + (size_t)MROWS * QKVN;          // 8192*3072

  auto cvt = [&](const float* src, unsigned short* dst, int n) {
    int n4 = n / 4;
    cvt_f32_bf16<<<(n4 + 255) / 256, 256, 0, stream>>>(src, dst, n4);
  };
  cvt(query, Xbf,    MROWS * DMODEL);
  cvt(w_qkv, Wqkvbf, QKVN  * DMODEL);
  cvt(w_out, Woutbf, DMODEL * DMODEL);

  // QKV projection: [8192,1024] x [1024,3072] -> bf16 [8192,3072]
  gemm_xwt<true, false><<<dim3(QKVN / 128, MROWS / 128), 256, 0, stream>>>(
      Xbf, Wqkvbf, nullptr, QKVbf, MROWS, QKVN, DMODEL);

  // attention: grid = (q blocks, heads, batch)
  attn_kernel<<<dim3(SEQ / 128, NHEAD, BATCH), 256, 0, stream>>>(QKVbf, AObf);

  // output projection + bias: [8192,1024] x [1024,1024] -> f32 d_out
  gemm_xwt<false, true><<<dim3(DMODEL / 128, MROWS / 128), 256, 0, stream>>>(
      AObf, Woutbf, b_out, out, MROWS, DMODEL, DMODEL);
}
